// TransfomerAttentionLayer_17171279250051
// MI455X (gfx1250) — matrix operand order repaired and checked by reading, not compiled, not run on hardware
//
#include <hip/hip_runtime.h>
#include <hip/hip_bf16.h>

// ---------------------------------------------------------------------------
// TGAT attention layer for gfx1250 (MI455X), wave32 + WMMA f16 path.
// Round 2: fragment-packed weights (coalesced B loads), LDS-staged outputs,
// vectorized fills and score loads.
// ---------------------------------------------------------------------------

typedef __attribute__((ext_vector_type(16))) _Float16 v16h;
typedef __attribute__((ext_vector_type(8)))  float    v8f;

#define DIM_NODE 128
#define DIM_EDGE 128
#define DIM_TIME 100
#define DIM_OUT  128
#define NUM_HEAD 8
#define DQ       228   // DIM_NODE + DIM_TIME
#define DQ_PAD   256
#define DKV      356   // DIM_NODE + DIM_EDGE + DIM_TIME
#define DKV_PAD  384
#define LN_EPS   1e-5f
#define NBLK     8     // 128 / 16 output-column blocks

// ---- WMMA A fragment from LDS (two contiguous 8-half runs -> 2x ds_load_b128)
__device__ inline v16h load_a_frag(const _Float16* A, int lda, int kb, int lane) {
    int r  = lane & 15;
    int hi = lane >> 4;
    v16h a;
#pragma unroll
    for (int j = 0; j < 16; ++j) {
        int k = kb + (j & 7) + hi * 8 + ((j >= 8) ? 16 : 0);
        a[j] = A[r * lda + k];
    }
    return a;
}

// ---- WMMA B fragment from fragment-packed weights: one v16h per lane -------
// Packed layout: frag[((kb32*NBLK + nb)*32 + lane)*16 + j]
__device__ inline v16h load_b_fragp(const _Float16* __restrict__ Wf, int kb32, int nb, int lane) {
    const v16h* p = (const v16h*)Wf;
    return p[(kb32 * NBLK + nb) * 32 + lane];
}

// ---- pack 4 floats -> 4 f16 -> single 8-byte LDS store ---------------------
__device__ inline void store4h(_Float16* dst, float a, float b, float c, float d) {
    _Float16 h[4] = {(_Float16)a, (_Float16)b, (_Float16)c, (_Float16)d};
    *reinterpret_cast<uint2*>(dst) = *reinterpret_cast<const uint2*>(h);
}

// ---- ordered-uint encoding for float atomicMax -----------------------------
__device__ inline unsigned enc_f32(float f) {
    unsigned u = __float_as_uint(f);
    return (u & 0x80000000u) ? ~u : (u | 0x80000000u);
}
__device__ inline float dec_f32(unsigned u) {
    unsigned b = (u & 0x80000000u) ? (u ^ 0x80000000u) : ~u;
    return __uint_as_float(b);
}

// ---------------------------------------------------------------------------
// K1: pack all four weight matrices into WMMA-B fragment layout (f16).
// idx -> {j[3:0], lane[8:4], nb[11:9], kb32[...]}; k = kb32*32 + (lane>=16)*8 + (j&7) + (j>=8)*16
// grid = 192 x 256 covers KPAD=384; 256-pad arrays guarded by kb32 < 8.
// ---------------------------------------------------------------------------
__global__ void prep_frag_kernel(const float* __restrict__ wq,
                                 const float* __restrict__ wk,
                                 const float* __restrict__ wv,
                                 const float* __restrict__ wout,
                                 _Float16* __restrict__ wq_f,
                                 _Float16* __restrict__ wk_f,
                                 _Float16* __restrict__ wv_f,
                                 _Float16* __restrict__ wout_f) {
    int idx  = blockIdx.x * 256 + threadIdx.x;      // < 12*4096 = 49152
    int j    = idx & 15;
    int lane = (idx >> 4) & 31;
    int nb   = (idx >> 9) & 7;
    int kb32 = idx >> 12;
    int k = kb32 * 32 + (lane >> 4) * 8 + (j & 7) + ((j >= 8) ? 16 : 0);
    int n = nb * 16 + (lane & 15);

    wk_f[idx] = (_Float16)((k < DKV) ? wk[n * DKV + k] : 0.0f);
    wv_f[idx] = (_Float16)((k < DKV) ? wv[n * DKV + k] : 0.0f);
    if (kb32 < 8) {
        wq_f[idx]   = (_Float16)((k < DQ) ? wq[n * DQ + k] : 0.0f);
        wout_f[idx] = (_Float16)wout[n * (DIM_NODE + DIM_OUT) + k];
    }
}

// ---------------------------------------------------------------------------
// K2: Qd = [q_data | cos(time_b) | 0] @ wq^T + wq_b   (16-node tiles, WMMA)
// ---------------------------------------------------------------------------
__global__ __launch_bounds__(256) void q_gemm_kernel(const float* __restrict__ src_feat,
                                                     const long long* __restrict__ src_idx,
                                                     const float* __restrict__ time_b,
                                                     const _Float16* __restrict__ wq_f,
                                                     const float* __restrict__ wq_b,
                                                     float* __restrict__ Qd, int D) {
    __shared__ _Float16 qin[16 * DQ_PAD];
    __shared__ float    qout[16 * DIM_OUT];
    int tile = blockIdx.x;
    int t    = threadIdx.x;

    // vectorized fill: 16 rows x 64 float4-chunks
    for (int idx = t; idx < 16 * (DQ_PAD / 4); idx += 256) {
        int r = idx >> 6, c = (idx & 63) * 4;
        int node = tile * 16 + r;
        float v0 = 0, v1 = 0, v2 = 0, v3 = 0;
        if (node < D) {
            if (c < DIM_NODE) {
                float4 f = *(const float4*)(src_feat + (long long)src_idx[node] * DIM_NODE + c);
                v0 = f.x; v1 = f.y; v2 = f.z; v3 = f.w;
            } else if (c < DQ) {   // DQ=228, multiple of 4
                int j = c - DIM_NODE;
                v0 = cosf(time_b[j]); v1 = cosf(time_b[j + 1]);
                v2 = cosf(time_b[j + 2]); v3 = cosf(time_b[j + 3]);
            }
        }
        store4h(&qin[r * DQ_PAD + c], v0, v1, v2, v3);
    }
    __syncthreads();

    int wave = t >> 5, lane = t & 31;
    v8f acc = {};
#pragma unroll
    for (int kb32 = 0; kb32 < DQ_PAD / 32; ++kb32) {
        v16h a = load_a_frag(qin, DQ_PAD, kb32 * 32, lane);
        v16h b = load_b_fragp(wq_f, kb32, wave, lane);
        acc = __builtin_amdgcn_wmma_f32_16x16x32_f16(false, a, false, b, (short)0, acc, false, false);
    }
    int n  = wave * 16 + (lane & 15);
    int hi = lane >> 4;
#pragma unroll
    for (int rr = 0; rr < 8; ++rr)
        qout[(rr + hi * 8) * DIM_OUT + n] = acc[rr] + wq_b[n];
    __syncthreads();

    // coalesced float4 store
    for (int idx = t; idx < 16 * (DIM_OUT / 4); idx += 256) {
        int r = idx >> 5, c = (idx & 31) * 4;
        int node = tile * 16 + r;
        if (node < D)
            *(float4*)(Qd + (long long)node * DIM_OUT + c) = *(const float4*)(qout + r * DIM_OUT + c);
    }
}

// ---------------------------------------------------------------------------
// K3: K/V = [kv_data | edge_f | cos(dt*w+b) | 0] @ {wk,wv}^T + b  (f16 out)
// ---------------------------------------------------------------------------
__global__ __launch_bounds__(256) void kv_gemm_kernel(const float* __restrict__ src_feat,
                                                      const long long* __restrict__ src_idx,
                                                      const float* __restrict__ edge_f,
                                                      const float* __restrict__ edge_dt,
                                                      const float* __restrict__ time_w,
                                                      const float* __restrict__ time_b,
                                                      const _Float16* __restrict__ wk_f,
                                                      const float* __restrict__ wk_b,
                                                      const _Float16* __restrict__ wv_f,
                                                      const float* __restrict__ wv_b,
                                                      _Float16* __restrict__ Kbuf,
                                                      _Float16* __restrict__ Vbuf,
                                                      int D, int E) {
    __shared__ _Float16 kvin[16 * DKV_PAD];
    __shared__ _Float16 ktile[16 * DIM_OUT];
    __shared__ _Float16 vtile[16 * DIM_OUT];
    int tile = blockIdx.x;
    int t    = threadIdx.x;

    // vectorized fill: 16 rows x 96 float4-chunks (boundaries 128/256/356 all %4==0)
    for (int idx = t; idx < 16 * (DKV_PAD / 4); idx += 256) {
        int r = idx / (DKV_PAD / 4), c = (idx % (DKV_PAD / 4)) * 4;
        int e = tile * 16 + r;
        float v0 = 0, v1 = 0, v2 = 0, v3 = 0;
        if (e < E) {
            if (c < DIM_NODE) {
                float4 f = *(const float4*)(src_feat + (long long)src_idx[D + e] * DIM_NODE + c);
                v0 = f.x; v1 = f.y; v2 = f.z; v3 = f.w;
            } else if (c < DIM_NODE + DIM_EDGE) {
                float4 f = *(const float4*)(edge_f + (long long)e * DIM_EDGE + (c - DIM_NODE));
                v0 = f.x; v1 = f.y; v2 = f.z; v3 = f.w;
            } else if (c < DKV) {  // DKV=356, multiple of 4
                int j = c - DIM_NODE - DIM_EDGE;
                float dt = edge_dt[e];
                v0 = cosf(dt * time_w[j]     + time_b[j]);
                v1 = cosf(dt * time_w[j + 1] + time_b[j + 1]);
                v2 = cosf(dt * time_w[j + 2] + time_b[j + 2]);
                v3 = cosf(dt * time_w[j + 3] + time_b[j + 3]);
            }
        }
        store4h(&kvin[r * DKV_PAD + c], v0, v1, v2, v3);
    }
    __syncthreads();

    int wave = t >> 5, lane = t & 31;
    v8f accK = {}, accV = {};
#pragma unroll
    for (int kb32 = 0; kb32 < DKV_PAD / 32; ++kb32) {
        v16h a  = load_a_frag(kvin, DKV_PAD, kb32 * 32, lane);
        v16h bk = load_b_fragp(wk_f, kb32, wave, lane);
        v16h bv = load_b_fragp(wv_f, kb32, wave, lane);
        accK = __builtin_amdgcn_wmma_f32_16x16x32_f16(false, a, false, bk, (short)0, accK, false, false);
        accV = __builtin_amdgcn_wmma_f32_16x16x32_f16(false, a, false, bv, (short)0, accV, false, false);
    }
    int n  = wave * 16 + (lane & 15);
    int hi = lane >> 4;
#pragma unroll
    for (int rr = 0; rr < 8; ++rr) {
        int m = rr + hi * 8;
        ktile[m * DIM_OUT + n] = (_Float16)(accK[rr] + wk_b[n]);
        vtile[m * DIM_OUT + n] = (_Float16)(accV[rr] + wv_b[n]);
    }
    __syncthreads();

    // coalesced b128 stores: 16 rows x 16 uint4-chunks = 256 per matrix
    {
        int r = t >> 4, q8 = (t & 15) * 8;
        int e = tile * 16 + r;
        if (e < E) {
            *(uint4*)(Kbuf + (long long)e * DIM_OUT + q8) = *(const uint4*)(ktile + r * DIM_OUT + q8);
            *(uint4*)(Vbuf + (long long)e * DIM_OUT + q8) = *(const uint4*)(vtile + r * DIM_OUT + q8);
        }
    }
}

// ---------------------------------------------------------------------------
// K4: init segment buffers
// ---------------------------------------------------------------------------
__global__ void init_seg_kernel(unsigned* __restrict__ m_enc,
                                float* __restrict__ den,
                                float* __restrict__ hbuf, int D) {
    int i = blockIdx.x * blockDim.x + threadIdx.x;
    if (i < D * DIM_OUT) hbuf[i] = 0.0f;
    if (i < D * NUM_HEAD) {
        m_enc[i] = 0x007FFFFFu;  // enc_f32(-inf)
        den[i]   = 0.0f;
    }
}

// ---------------------------------------------------------------------------
// K5: per-(edge,head) score = leaky_relu(Q.K), atomicMax segment max
// ---------------------------------------------------------------------------
__global__ void score_max_kernel(const float* __restrict__ Qd,
                                 const _Float16* __restrict__ Kbuf,
                                 const long long* __restrict__ edge_dst,
                                 float* __restrict__ scores,
                                 unsigned* __restrict__ m_enc, int E) {
    long long i = (long long)blockIdx.x * blockDim.x + threadIdx.x;
    if (i >= (long long)E * NUM_HEAD) return;
    int e = (int)(i >> 3), h = (int)(i & 7);
    long long dst = edge_dst[e];

    v16h kv = *(const v16h*)(Kbuf + (long long)e * DIM_OUT + h * 16);  // 32B aligned
    const float4* qp = (const float4*)(Qd + dst * DIM_OUT + h * 16);
    float s = 0.0f;
#pragma unroll
    for (int q = 0; q < 4; ++q) {
        float4 f = qp[q];
        s += f.x * (float)kv[q * 4 + 0] + f.y * (float)kv[q * 4 + 1]
           + f.z * (float)kv[q * 4 + 2] + f.w * (float)kv[q * 4 + 3];
    }
    s = (s > 0.0f) ? s : 0.2f * s;   // leaky_relu(0.2)
    scores[i] = s;
    atomicMax(&m_enc[dst * NUM_HEAD + h], enc_f32(s));
}

// ---------------------------------------------------------------------------
// K6: ex = exp(score - m), accumulate denominators
// ---------------------------------------------------------------------------
__global__ void exp_den_kernel(const long long* __restrict__ edge_dst,
                               const unsigned* __restrict__ m_enc,
                               float* __restrict__ scores,
                               float* __restrict__ den, int E) {
    long long i = (long long)blockIdx.x * blockDim.x + threadIdx.x;
    if (i >= (long long)E * NUM_HEAD) return;
    int e = (int)(i >> 3), h = (int)(i & 7);
    long long dst = edge_dst[e];
    float m = dec_f32(m_enc[dst * NUM_HEAD + h]);
    if (!isfinite(m)) m = 0.0f;
    float ex = expf(scores[i] - m);
    scores[i] = ex;
    atomicAdd(&den[dst * NUM_HEAD + h], ex);
}

// ---------------------------------------------------------------------------
// K7: h[dst] += att * V   (f32 scatter-add; edge_dst sorted -> hot L2 lines)
// ---------------------------------------------------------------------------
__global__ void agg_kernel(const _Float16* __restrict__ Vbuf,
                           const float* __restrict__ scores,
                           const float* __restrict__ den,
                           const long long* __restrict__ edge_dst,
                           float* __restrict__ hbuf, int E) {
    long long i = (long long)blockIdx.x * blockDim.x + threadIdx.x;
    if (i >= (long long)E * DIM_OUT) return;
    int e = (int)(i >> 7), c = (int)(i & 127);
    int h = c >> 4;
    long long dst = edge_dst[e];
    float att = scores[(long long)e * NUM_HEAD + h] / den[dst * NUM_HEAD + h];
    atomicAdd(&hbuf[dst * DIM_OUT + c], (float)Vbuf[i] * att);
}

// ---------------------------------------------------------------------------
// K8: rst = relu([h | q_data] @ wout^T + b); per-row LayerNorm
// ---------------------------------------------------------------------------
__global__ __launch_bounds__(256) void out_gemm_ln_kernel(const float* __restrict__ hbuf,
                                                          const float* __restrict__ src_feat,
                                                          const long long* __restrict__ src_idx,
                                                          const _Float16* __restrict__ wout_f,
                                                          const float* __restrict__ wout_b,
                                                          const float* __restrict__ ln_g,
                                                          const float* __restrict__ ln_b,
                                                          float* __restrict__ out, int D) {
    __shared__ _Float16 ain[16 * DQ_PAD];      // 16 x 256
    __shared__ float    otile[16 * DIM_OUT];   // 16 x 128
    __shared__ float    mu_s[16], inv_s[16];
    int tile = blockIdx.x;
    int t    = threadIdx.x;

    for (int idx = t; idx < 16 * (DQ_PAD / 4); idx += 256) {
        int r = idx >> 6, c = (idx & 63) * 4;
        int node = tile * 16 + r;
        float v0 = 0, v1 = 0, v2 = 0, v3 = 0;
        if (node < D) {
            float4 f;
            if (c < DIM_OUT)
                f = *(const float4*)(hbuf + (long long)node * DIM_OUT + c);
            else
                f = *(const float4*)(src_feat + (long long)src_idx[node] * DIM_NODE + (c - DIM_OUT));
            v0 = f.x; v1 = f.y; v2 = f.z; v3 = f.w;
        }
        store4h(&ain[r * DQ_PAD + c], v0, v1, v2, v3);
    }
    __syncthreads();

    int wave = t >> 5, lane = t & 31;
    v8f acc = {};
#pragma unroll
    for (int kb32 = 0; kb32 < DQ_PAD / 32; ++kb32) {
        v16h a = load_a_frag(ain, DQ_PAD, kb32 * 32, lane);
        v16h b = load_b_fragp(wout_f, kb32, wave, lane);
        acc = __builtin_amdgcn_wmma_f32_16x16x32_f16(false, a, false, b, (short)0, acc, false, false);
    }
    int n  = wave * 16 + (lane & 15);
    int hi = lane >> 4;
#pragma unroll
    for (int rr = 0; rr < 8; ++rr) {
        int m = rr + hi * 8;
        float r = acc[rr] + wout_b[n];
        otile[m * DIM_OUT + n] = (r > 0.0f) ? r : 0.0f;   // relu
    }
    __syncthreads();

    if (t < 16) {
        float mu = 0.0f;
        for (int c = 0; c < DIM_OUT; ++c) mu += otile[t * DIM_OUT + c];
        mu *= (1.0f / DIM_OUT);
        float var = 0.0f;
        for (int c = 0; c < DIM_OUT; ++c) {
            float d = otile[t * DIM_OUT + c] - mu;
            var += d * d;
        }
        var *= (1.0f / DIM_OUT);
        mu_s[t]  = mu;
        inv_s[t] = rsqrtf(var + LN_EPS);
    }
    __syncthreads();

    // coalesced float4 write-out with LN applied
    for (int idx = t; idx < 16 * (DIM_OUT / 4); idx += 256) {
        int r = idx >> 5, c = (idx & 31) * 4;
        int node = tile * 16 + r;
        if (node < D) {
            float mu = mu_s[r], inv = inv_s[r];
            float4 g = *(const float4*)(ln_g + c);
            float4 bb = *(const float4*)(ln_b + c);
            const float* o = otile + r * DIM_OUT + c;
            float4 res;
            res.x = (o[0] - mu) * inv * g.x + bb.x;
            res.y = (o[1] - mu) * inv * g.y + bb.y;
            res.z = (o[2] - mu) * inv * g.z + bb.z;
            res.w = (o[3] - mu) * inv * g.w + bb.w;
            *(float4*)(out + (long long)node * DIM_OUT + c) = res;
        }
    }
}

// ---------------------------------------------------------------------------
// Host entry
// ---------------------------------------------------------------------------
extern "C" void kernel_launch(void* const* d_in, const int* in_sizes, int n_in,
                              void* d_out, int out_size, void* d_ws, size_t ws_size,
                              hipStream_t stream) {
    const float*     src_feat = (const float*)d_in[0];
    const float*     edge_f   = (const float*)d_in[1];
    const float*     edge_dt  = (const float*)d_in[2];
    const float*     time_w   = (const float*)d_in[3];
    const float*     time_b   = (const float*)d_in[4];
    const float*     wq_w     = (const float*)d_in[5];
    const float*     wq_b     = (const float*)d_in[6];
    const float*     wk_w     = (const float*)d_in[7];
    const float*     wk_b     = (const float*)d_in[8];
    const float*     wv_w     = (const float*)d_in[9];
    const float*     wv_b     = (const float*)d_in[10];
    const float*     wout_w   = (const float*)d_in[11];
    const float*     wout_b   = (const float*)d_in[12];
    const float*     ln_g     = (const float*)d_in[13];
    const float*     ln_b     = (const float*)d_in[14];
    const long long* src_idx  = (const long long*)d_in[15];
    const long long* edge_dst = (const long long*)d_in[16];
    float* out = (float*)d_out;

    const int E = in_sizes[1] / DIM_EDGE;
    const int D = in_sizes[0] / DIM_NODE - E;

    char* p = (char*)d_ws;
    auto alloc = [&](size_t bytes) -> void* {
        void* r = (void*)p;
        p += (bytes + 255) & ~(size_t)255;
        return r;
    };
    _Float16* wq_f   = (_Float16*)alloc((size_t)DQ_PAD * DIM_OUT * 2);
    _Float16* wk_f   = (_Float16*)alloc((size_t)DKV_PAD * DIM_OUT * 2);
    _Float16* wv_f   = (_Float16*)alloc((size_t)DKV_PAD * DIM_OUT * 2);
    _Float16* wout_f = (_Float16*)alloc((size_t)DQ_PAD * DIM_OUT * 2);
    float*    Qd     = (float*)alloc((size_t)D * DIM_OUT * 4);
    _Float16* Kbuf   = (_Float16*)alloc((size_t)E * DIM_OUT * 2);
    _Float16* Vbuf   = (_Float16*)alloc((size_t)E * DIM_OUT * 2);
    float*    scores = (float*)alloc((size_t)E * NUM_HEAD * 4);
    unsigned* m_enc  = (unsigned*)alloc((size_t)D * NUM_HEAD * 4);
    float*    den    = (float*)alloc((size_t)D * NUM_HEAD * 4);
    float*    hbuf   = (float*)alloc((size_t)D * DIM_OUT * 4);

    const int dTiles = (D + 15) / 16;
    const int eTiles = (E + 15) / 16;

    prep_frag_kernel<<<(DKV_PAD / 32) * 4096 / 256, 256, 0, stream>>>(
        wq_w, wk_w, wv_w, wout_w, wq_f, wk_f, wv_f, wout_f);

    q_gemm_kernel<<<dTiles, 256, 0, stream>>>(
        src_feat, src_idx, time_b, wq_f, wq_b, Qd, D);

    kv_gemm_kernel<<<eTiles, 256, 0, stream>>>(
        src_feat, src_idx, edge_f, edge_dt, time_w, time_b,
        wk_f, wk_b, wv_f, wv_b, Kbuf, Vbuf, D, E);

    {
        int n = D * DIM_OUT;
        init_seg_kernel<<<(n + 255) / 256, 256, 0, stream>>>(m_enc, den, hbuf, D);
    }
    {
        long long n = (long long)E * NUM_HEAD;
        score_max_kernel<<<(unsigned)((n + 255) / 256), 256, 0, stream>>>(
            Qd, Kbuf, edge_dst, scores, m_enc, E);
        exp_den_kernel<<<(unsigned)((n + 255) / 256), 256, 0, stream>>>(
            edge_dst, m_enc, scores, den, E);
    }
    {
        long long n = (long long)E * DIM_OUT;
        agg_kernel<<<(unsigned)((n + 255) / 256), 256, 0, stream>>>(
            Vbuf, scores, den, edge_dst, hbuf, E);
    }

    out_gemm_ln_kernel<<<dTiles, 256, 0, stream>>>(
        hbuf, src_feat, src_idx, wout_f, wout_b, ln_g, ln_b, out, D);
}
